// GNNML3Model_84086869721474
// MI455X (gfx1250) — matrix operand order repaired and checked by reading, not compiled, not run on hardware
//
#include <hip/hip_runtime.h>
#include <math.h>

#define N_NODES 50000
#define N_EDGES 800000
#define DIM_IN  32
#define DIM_H   128
#define DIM_OUT 128
#define K_SUP   5
#define H2      64
#define KCAT    (K_SUP * DIM_H)   // 640

typedef __attribute__((ext_vector_type(16))) _Float16 v16h;
typedef __attribute__((ext_vector_type(8)))  float    v8f;

// ---------------------------------------------------------------------------
// Weight f32 -> f16 conversion with WMMA B-fragment swizzle.
// Output layout: out[((nt*(K/32)+kc)*32 + lane)*16 + j]
//   k = kc*32 + (j&7) + 8*(lane>>4) + 16*(j>>3),  n = nt*16 + (lane&15)
// (B 32x16 f16 layout: lanes 0-15 hold K=0..7,16..23, lanes 16-31 hold K=8..15,24..31)
// ---------------------------------------------------------------------------
__global__ __launch_bounds__(256)
void k_swizzleB(const float* __restrict__ W, _Float16* __restrict__ out,
                int K, int N)
{
    int o = blockIdx.x * 256 + threadIdx.x;
    if (o >= K * N) return;
    int kchunks = K >> 5;
    int j    = o & 15;
    int lane = (o >> 4) & 31;
    int cl   = o >> 9;                 // nt*kchunks + kc
    int kc   = cl % kchunks;
    int nt   = cl / kchunks;
    int k = kc * 32 + (j & 7) + ((lane >> 4) << 3) + ((j >> 3) << 4);
    int n = nt * 16 + (lane & 15);
    out[o] = (_Float16)W[k * N + n];
}

// Same, but B = [fc11_w | fc12_w] concatenated along columns (K=128, N=128).
__global__ __launch_bounds__(256)
void k_swizzleB_pair(const float* __restrict__ W1, const float* __restrict__ W2,
                     _Float16* __restrict__ out)
{
    const int K = DIM_H, N = DIM_H;
    int o = blockIdx.x * 256 + threadIdx.x;
    if (o >= K * N) return;
    int kchunks = K >> 5;              // 4
    int j    = o & 15;
    int lane = (o >> 4) & 31;
    int cl   = o >> 9;
    int kc   = cl % kchunks;
    int nt   = cl / kchunks;
    int k = kc * 32 + (j & 7) + ((lane >> 4) << 3) + ((j >> 3) << 4);
    int n = nt * 16 + (lane & 15);
    float v = (n < H2) ? W1[k * H2 + n] : W2[k * H2 + (n - H2)];
    out[o] = (_Float16)v;
}

// ---------------------------------------------------------------------------
// Generic tall GEMM: C[m, colOff+n] = act(A(AT, MxK) @ Bsw(f16, KxN) + bias)
// One wave per 16x16 tile; 8 waves per block. Bsw is pre-swizzled.
// act: 0 = none, 1 = relu. useLds: stage B in LDS (dyn-shared = K*N*2 bytes).
// Distinct K-loops per branch keep LDS reads on ds_load and global reads on
// global_load (no generic/flat fallback).
// ---------------------------------------------------------------------------
#define GEMM_KLOOP(BPTR)                                                      \
    for (int kc = 0; kc < kChunks; ++kc) {                                    \
        const AT* g0 = Arow + kc * 32;                                        \
        v16h a;                                                               \
        _Pragma("unroll")                                                     \
        for (int j = 0; j < 8; ++j) {                                         \
            a[j]     = (_Float16)g0[j];                                       \
            a[j + 8] = (_Float16)g0[16 + j];                                  \
        }                                                                     \
        v16h b = *(const v16h*)((BPTR) +                                      \
                                (((nt * kChunks + kc) * 32 + lane) << 4));    \
        acc = __builtin_amdgcn_wmma_f32_16x16x32_f16(                         \
            false, a, false, b, (short)0, acc, false, false);                 \
    }

template <typename AT, typename CT>
__global__ __launch_bounds__(256)
void k_gemm_wmma(const AT* __restrict__ A, int lda,
                 const _Float16* __restrict__ Bsw,
                 const float* __restrict__ bias,
                 CT* __restrict__ C, int ldc, int colOff,
                 int Mtiles, int K, int N, int act, int useLds)
{
    extern __shared__ __align__(32) _Float16 Bs[];
    const int tid  = threadIdx.x;
    const int lane = tid & 31;
    const int w    = tid >> 5;
    const int nTiles  = N >> 4;
    const int mPerBlk = 8 / nTiles;
    const int kChunks = K >> 5;

    const int mt0 = blockIdx.x * mPerBlk + w / nTiles;
    const int nt  = w % nTiles;
    const int l16 = lane >> 4;
    const AT* Arow = A + (long)(mt0 * 16 + (lane & 15)) * lda + l16 * 8;
    if (mt0 < Mtiles) __builtin_prefetch(Arow, 0, 0);   // global_prefetch_b8

    if (useLds) {
        const unsigned* s = (const unsigned*)Bsw;
        unsigned* d = (unsigned*)Bs;
        int cnt = (K * N) >> 1;
        for (int i = tid; i < cnt; i += 256) d[i] = s[i];
        __syncthreads();
    }
    if (mt0 >= Mtiles) return;

    v8f acc = {};
    if (useLds) {
        GEMM_KLOOP(Bs)          // ds_load_b128 path
    } else {
        GEMM_KLOOP(Bsw)         // global_load_b128 path
    }

    const int n  = nt * 16 + (lane & 15);
    const float bv = bias ? bias[n] : 0.0f;
#pragma unroll
    for (int r = 0; r < 8; ++r) {
        int m = mt0 * 16 + r + (l16 << 3);
        float v = acc[r] + bv;
        if (act) v = fmaxf(v, 0.0f);
        C[(long)m * ldc + colOff + n] = (CT)v;
    }
}

// ---------------------------------------------------------------------------
// em gate: y = A(f16, Mx128) @ [fc11|fc12](128x128)
// out[:,64+n] = tanh(y1+b1)*tanh(y2+b2); each wave does paired tiles nt, nt+4.
// ---------------------------------------------------------------------------
__global__ __launch_bounds__(256)
void k_em_wmma(const _Float16* __restrict__ A, const _Float16* __restrict__ Bsw,
               const float* __restrict__ b1, const float* __restrict__ b2,
               _Float16* __restrict__ C, int Mtiles)
{
    __shared__ __align__(32) _Float16 Bsl[DIM_H * DIM_H];
    const int tid  = threadIdx.x;
    const int lane = tid & 31;
    const int w    = tid >> 5;
    const int mt = blockIdx.x * 2 + (w >> 2);
    const int nt = w & 3;
    const int l16 = lane >> 4;
    const _Float16* Arow =
        A + (long)(mt * 16 + (lane & 15)) * DIM_H + l16 * 8;
    if (mt < Mtiles) __builtin_prefetch(Arow, 0, 0);
    {
        const unsigned* s = (const unsigned*)Bsw;
        unsigned* d = (unsigned*)Bsl;
        for (int i = tid; i < (DIM_H * DIM_H) / 2; i += 256) d[i] = s[i];
    }
    __syncthreads();
    if (mt >= Mtiles) return;

    const int kChunks = DIM_H / 32;    // 4
    v8f c1 = {}, c2 = {};
    for (int kc = 0; kc < kChunks; ++kc) {
        const _Float16* g0 = Arow + kc * 32;
        v16h a;
#pragma unroll
        for (int j = 0; j < 8; ++j) {
            a[j]     = g0[j];
            a[j + 8] = g0[16 + j];
        }
        v16h bb1 = *(const v16h*)(Bsl + ((( nt      * kChunks + kc) * 32 + lane) << 4));
        v16h bb2 = *(const v16h*)(Bsl + ((((nt + 4) * kChunks + kc) * 32 + lane) << 4));
        c1 = __builtin_amdgcn_wmma_f32_16x16x32_f16(false, a, false, bb1, (short)0, c1, false, false);
        c2 = __builtin_amdgcn_wmma_f32_16x16x32_f16(false, a, false, bb2, (short)0, c2, false, false);
    }

    const int n   = lane & 15;
    const float bv1 = b1[nt * 16 + n];
    const float bv2 = b2[nt * 16 + n];
#pragma unroll
    for (int r = 0; r < 8; ++r) {
        int m = mt * 16 + r + (l16 << 3);
        float v = tanhf(c1[r] + bv1) * tanhf(c2[r] + bv2);
        C[(long)m * DIM_H + H2 + nt * 16 + n] = (_Float16)v;
    }
}

// ---------------------------------------------------------------------------
// Edge MLP: ea = relu(cat[relu(a@e1), tanh(a@e2)*tanh(a@e3)] @ e4)
// ---------------------------------------------------------------------------
__global__ __launch_bounds__(256)
void k_edge_mlp(const float* __restrict__ attr,
                const float* __restrict__ e1, const float* __restrict__ e2,
                const float* __restrict__ e3, const float* __restrict__ e4,
                float* __restrict__ ea_out)
{
    int e = blockIdx.x * 256 + threadIdx.x;
    if (e >= N_EDGES) return;
    float a[K_SUP];
#pragma unroll
    for (int i = 0; i < K_SUP; ++i) a[i] = attr[e * K_SUP + i];

    float cat[4 * K_SUP];
#pragma unroll
    for (int j = 0; j < 2 * K_SUP; ++j) {
        float s1 = 0.f, s2 = 0.f, s3 = 0.f;
#pragma unroll
        for (int i = 0; i < K_SUP; ++i) {
            s1 += a[i] * e1[i * (2 * K_SUP) + j];
            s2 += a[i] * e2[i * (2 * K_SUP) + j];
            s3 += a[i] * e3[i * (2 * K_SUP) + j];
        }
        cat[j]             = fmaxf(s1, 0.f);
        cat[2 * K_SUP + j] = tanhf(s2) * tanhf(s3);
    }
#pragma unroll
    for (int i = 0; i < K_SUP; ++i) {
        float s = 0.f;
#pragma unroll
        for (int j = 0; j < 4 * K_SUP; ++j) s += cat[j] * e4[j * K_SUP + i];
        ea_out[e * K_SUP + i] = fmaxf(s, 0.f);
    }
}

// ---------------------------------------------------------------------------
// Fused scatter: acc[dst, i*128+f] += ea[e,i] * (float)H[src, f]   (i = 0..4)
// 128 threads (features) per edge; H is f16 (halves L2 gather bytes);
// native f32 global atomics (L2-resident accumulator).
// ---------------------------------------------------------------------------
__global__ __launch_bounds__(256)
void k_scatter(const _Float16* __restrict__ H, const long long* __restrict__ ei,
               const float* __restrict__ ea, float* __restrict__ acc)
{
    long t = (long)blockIdx.x * 256 + threadIdx.x;
    int e = (int)(t >> 7);
    int f = (int)(t & 127);
    if (e >= N_EDGES) return;
    int src = (int)ei[e];
    int dst = (int)ei[N_EDGES + e];
    float xv = (float)H[(long)src * DIM_H + f];
    const float* eaP = ea + (long)e * K_SUP;
    float* base = acc + (long)dst * KCAT + f;
#pragma unroll
    for (int i = 0; i < K_SUP; ++i) {
        __hip_atomic_fetch_add(base + i * DIM_H, eaP[i] * xv,
                               __ATOMIC_RELAXED, __HIP_MEMORY_SCOPE_AGENT);
    }
}

// ---------------------------------------------------------------------------
extern "C" void kernel_launch(void* const* d_in, const int* in_sizes, int n_in,
                              void* d_out, int out_size, void* d_ws, size_t ws_size,
                              hipStream_t stream)
{
    // setup_inputs() dict order, params flattened in insertion order
    const float*     x     = (const float*)d_in[0];
    const long long* ei    = (const long long*)d_in[1];
    const float*     eattr = (const float*)d_in[2];
    const float*     in_w  = (const float*)d_in[3];
    const float*     in_b  = (const float*)d_in[4];
    const float*     out_w = (const float*)d_in[5];
    const float*     out_b = (const float*)d_in[6];
    // layers: 7 + l*10 : e1,e2,e3,e4,W,b,fc11_w,fc11_b,fc12_w,fc12_b

    // workspace carve (~170 MB)
    float*    acc = (float*)d_ws;                               // 50000*640 f32
    float*    ea  = acc + (size_t)N_NODES * KCAT;               // 800000*5 f32
    _Float16* hA  = (_Float16*)(ea + (size_t)N_EDGES * K_SUP);  // 50000*128 f16
    _Float16* hB  = hA + (size_t)N_NODES * DIM_H;               // 50000*128 f16
    _Float16* wh  = hB + (size_t)N_NODES * DIM_H;               // 40960 f16 (reused)

    const int MT  = N_NODES / 16;       // 3125 (exact)
    const int MT2 = (MT + 1) / 2;       // 1563

    // input projection: hA = (f16)(x @ in_w + in_b)
    k_swizzleB<<<(DIM_IN * DIM_H + 255) / 256, 256, 0, stream>>>(in_w, wh, DIM_IN, DIM_H);
    k_gemm_wmma<float, _Float16><<<MT, 256, DIM_IN * DIM_H * 2, stream>>>(
        x, DIM_IN, wh, in_b, hA, DIM_H, 0, MT, DIM_IN, DIM_H, 0, 1);

    _Float16* cur = hA;
    _Float16* nxt = hB;
    for (int l = 0; l < 3; ++l) {
        const int base = 7 + l * 10;
        const float* e1     = (const float*)d_in[base + 0];
        const float* e2     = (const float*)d_in[base + 1];
        const float* e3     = (const float*)d_in[base + 2];
        const float* e4     = (const float*)d_in[base + 3];
        const float* W      = (const float*)d_in[base + 4];   // [5][128][64] = 640x64
        const float* b      = (const float*)d_in[base + 5];
        const float* fc11_w = (const float*)d_in[base + 6];
        const float* fc11_b = (const float*)d_in[base + 7];
        const float* fc12_w = (const float*)d_in[base + 8];
        const float* fc12_b = (const float*)d_in[base + 9];

        // edge MLP -> ea
        k_edge_mlp<<<N_EDGES / 256, 256, 0, stream>>>(eattr, e1, e2, e3, e4, ea);

        // zero accumulator, scatter-add messages
        hipMemsetAsync(acc, 0, (size_t)N_NODES * KCAT * sizeof(float), stream);
        k_scatter<<<(N_EDGES * 128) / 256, 256, 0, stream>>>(cur, ei, ea, acc);

        // spec = relu(acc @ W + b) -> nxt[:, 0:64]   (B = 80KB, L2-cached global)
        k_swizzleB<<<(KCAT * H2 + 255) / 256, 256, 0, stream>>>(W, wh, KCAT, H2);
        k_gemm_wmma<float, _Float16><<<MT2, 256, 0, stream>>>(
            acc, KCAT, wh, b, nxt, DIM_H, 0, MT, KCAT, H2, 1, 0);

        // em = tanh(cur@fc11+b11)*tanh(cur@fc12+b12) -> nxt[:, 64:128]
        k_swizzleB_pair<<<(DIM_H * DIM_H + 255) / 256, 256, 0, stream>>>(fc11_w, fc12_w, wh);
        k_em_wmma<<<MT2, 256, 0, stream>>>(cur, wh, fc11_b, fc12_b, nxt, MT);

        _Float16* t = cur; cur = nxt; nxt = t;
    }

    // output projection: d_out = cur @ out_w + out_b (f32 out)
    k_swizzleB<<<(DIM_H * DIM_OUT + 255) / 256, 256, 0, stream>>>(out_w, wh, DIM_H, DIM_OUT);
    k_gemm_wmma<_Float16, float><<<MT, 256, DIM_H * DIM_OUT * 2, stream>>>(
        cur, DIM_H, wh, out_b, (float*)d_out, DIM_OUT, 0, MT, DIM_H, DIM_OUT, 0, 1);
}
